// STAttention_74947179316138
// MI455X (gfx1250) — matrix operand order repaired and checked by reading, not compiled, not run on hardware
//
#include <hip/hip_runtime.h>
#include <math.h>

// ---------------- problem constants (match reference) ----------------
constexpr int cB = 4, cT = 12, cN = 512, cD = 256, cH = 4;
constexpr int cHD = 16, cHDS = 8, cSEG = 6, cC0 = 64, cC1 = 128;
constexpr int MTOK = cB * cT * cN;          // 24576 tokens of x
constexpr int GG   = cB * cT * cH;          // 192 (b,t,h) groups
constexpr float LAMBDA_INIT = 0.2f;

typedef __attribute__((ext_vector_type(16))) _Float16 v16h;
typedef __attribute__((ext_vector_type(8)))  _Float16 v8h;
typedef __attribute__((ext_vector_type(8)))  float    v8f;

// ---------------- WMMA wrapper ----------------
__device__ __forceinline__ v8f wmma_f16(v16h a, v16h b, v8f c) {
  return __builtin_amdgcn_wmma_f32_16x16x32_f16(
      false, a, false, b, (short)0, c, false, false);
}

// pack two float4 into 8 consecutive f16 lanes of a fragment (scaled)
__device__ __forceinline__ void cvt8(v16h& a, int base, float4 x0, float4 x1, float s) {
  a[base + 0] = (_Float16)(x0.x * s); a[base + 1] = (_Float16)(x0.y * s);
  a[base + 2] = (_Float16)(x0.z * s); a[base + 3] = (_Float16)(x0.w * s);
  a[base + 4] = (_Float16)(x1.x * s); a[base + 5] = (_Float16)(x1.y * s);
  a[base + 6] = (_Float16)(x1.z * s); a[base + 7] = (_Float16)(x1.w * s);
}
__device__ __forceinline__ void zero8(v16h& a, int base) {
#pragma unroll
  for (int j = 0; j < 8; ++j) a[base + j] = (_Float16)0.0f;
}

// ============ native-f16 fragment loaders (b128 loads, no conversion) ============
// A 16x32 f16: lane&15=row M; lanes<16 K {0..7,16..23}; lanes>=16 K {8..15,24..31}
__device__ __forceinline__ v16h afrag_h(const _Float16* __restrict__ p, int ld,
                                        int mbase, int kbase, int lane) {
  int m  = mbase + (lane & 15);
  int lo = (lane < 16) ? 0 : 8;
  const _Float16* row = p + (size_t)m * (size_t)ld + kbase + lo;
  v8h x0 = *reinterpret_cast<const v8h*>(row);        // 8 halves = one b128
  v8h x1 = *reinterpret_cast<const v8h*>(row + 16);
  return __builtin_shufflevector(x0, x1, 0, 1, 2, 3, 4, 5, 6, 7,
                                 8, 9, 10, 11, 12, 13, 14, 15);
}
// B 32x16 f16: lane&15=col N; lanes<16 K {0..7,16..23}; lanes>=16 K {8..15,24..31}
__device__ __forceinline__ v16h bfrag_h(const _Float16* __restrict__ p, int ld,
                                        int nbase, int kbase, int lane) {
  int n  = nbase + (lane & 15);
  int kb = (lane < 16) ? 0 : 8;
  const _Float16* row = p + (size_t)n * (size_t)ld + kbase + kb;
  v8h x0 = *reinterpret_cast<const v8h*>(row);
  v8h x1 = *reinterpret_cast<const v8h*>(row + 16);
  return __builtin_shufflevector(x0, x1, 0, 1, 2, 3, 4, 5, 6, 7,
                                 8, 9, 10, 11, 12, 13, 14, 15);
}

// ============ f32-source loaders used by attention kernels ============
// A 16x32 from f32 rows, unguarded, vector loads
__device__ __forceinline__ v16h afrag_rows_v(const float* __restrict__ p, int ld,
                                             int mbase, int kbase, int lane) {
  int m  = mbase + (lane & 15);
  int lo = (lane < 16) ? 0 : 8;
  const float* row = p + (size_t)m * (size_t)ld + kbase + lo;
  float4 x0 = *reinterpret_cast<const float4*>(row);
  float4 x1 = *reinterpret_cast<const float4*>(row + 4);
  float4 x2 = *reinterpret_cast<const float4*>(row + 16);
  float4 x3 = *reinterpret_cast<const float4*>(row + 20);
  v16h a;
  cvt8(a, 0, x0, x1, 1.0f);
  cvt8(a, 8, x2, x3, 1.0f);
  return a;
}
// K=8 specializations (diff-attn half heads)
__device__ __forceinline__ v16h afrag_k8_v(const float* __restrict__ p, int ld,
                                           int mbase, int lane) {
  int m = mbase + (lane & 15);
  const float* row = p + (size_t)m * (size_t)ld;
  float4 x0 = *reinterpret_cast<const float4*>(row);
  float4 x1 = *reinterpret_cast<const float4*>(row + 4);
  float s = (lane < 16) ? 1.0f : 0.0f;
  v16h a;
  cvt8(a, 0, x0, x1, s);
  zero8(a, 8);
  return a;
}
__device__ __forceinline__ v16h bfrag_k8_v(const float* __restrict__ p, int ld,
                                           int nbase, int lane) {
  int n = nbase + (lane & 15);
  const float* row = p + (size_t)n * (size_t)ld;
  float4 x0 = *reinterpret_cast<const float4*>(row);
  float4 x1 = *reinterpret_cast<const float4*>(row + 4);
  float s = (lane < 16) ? 1.0f : 0.0f;
  v16h b;
  cvt8(b, 0, x0, x1, s);
  zero8(b, 8);
  return b;
}
// K=16 specializations (HD=16 heads), row clamp+mask
__device__ __forceinline__ v16h afrag_k16_v(const float* __restrict__ p, int ld,
                                            int mbase, int mvalid, int lane) {
  int m = mbase + (lane & 15);
  bool mok = (m < mvalid);
  int mc = mok ? m : mbase;
  int lo = (lane < 16) ? 0 : 8;
  const float* row = p + (size_t)mc * (size_t)ld + lo;
  float4 x0 = *reinterpret_cast<const float4*>(row);
  float4 x1 = *reinterpret_cast<const float4*>(row + 4);
  v16h a;
  cvt8(a, 0, x0, x1, mok ? 1.0f : 0.0f);
  zero8(a, 8);
  return a;
}
__device__ __forceinline__ v16h bfrag_k16_v(const float* __restrict__ p, int ld,
                                            int nbase, int nvalid, int lane) {
  int n = nbase + (lane & 15);
  bool nok = (n < nvalid);
  int nc = nok ? n : nbase;
  int lo = (lane < 16) ? 0 : 8;
  const float* row = p + (size_t)nc * (size_t)ld + lo;
  float4 x0 = *reinterpret_cast<const float4*>(row);
  float4 x1 = *reinterpret_cast<const float4*>(row + 4);
  float s = nok ? 1.0f : 0.0f;
  v16h b;
  cvt8(b, 0, x0, x1, s);
  zero8(b, 8);
  return b;
}
// strided loaders (scalar, branch-free)
__device__ __forceinline__ v16h afrag_cols_v(const float* __restrict__ p, int ld,
                                             int mbase, int kbase, int lane) {
  int m  = mbase + (lane & 15);
  int lo = (lane < 16) ? 0 : 8;
  v16h a;
#pragma unroll
  for (int j = 0; j < 8; ++j) {
    a[j]     = (_Float16)p[(size_t)(kbase + lo + j) * ld + m];
    a[j + 8] = (_Float16)p[(size_t)(kbase + 16 + lo + j) * ld + m];
  }
  return a;
}
__device__ __forceinline__ v16h bfrag_cols_v(const float* __restrict__ p, int ld,
                                             int kbase, int nbase, int lane) {
  int n  = nbase + (lane & 15);
  int lo = (lane < 16) ? 0 : 8;
  v16h b;
#pragma unroll
  for (int j = 0; j < 8; ++j) {
    b[j]     = (_Float16)p[(size_t)(kbase + lo + j) * ld + n];
    b[j + 8] = (_Float16)p[(size_t)(kbase + 16 + lo + j) * ld + n];
  }
  return b;
}
__device__ __forceinline__ v16h bfrag_cols_g(const float* __restrict__ p, int ld,
                                             int kbase, int nbase, int kvalid, int lane) {
  int n  = nbase + (lane & 15);
  int lo = (lane < 16) ? 0 : 8;
  v16h b;
#pragma unroll
  for (int j = 0; j < 16; ++j) {
    int k = (j < 8) ? (lo + j) : (16 + lo + (j - 8));
    bool ok = (k < kvalid);
    float f = p[(size_t)(kbase + (ok ? k : 0)) * ld + n];
    b[j] = (_Float16)(ok ? f : 0.0f);
  }
  return b;
}
__device__ __forceinline__ v16h afrag_rows_g(const float* __restrict__ p, int ld,
                                             int mbase, int kbase,
                                             int mvalid, int kvalid, int lane) {
  int m = mbase + (lane & 15);
  bool mok = (m < mvalid);
  int mc = mok ? m : mbase;
  int lo = (lane < 16) ? 0 : 8;
  const float* row = p + (size_t)mc * (size_t)ld + kbase;
  v16h a;
#pragma unroll
  for (int j = 0; j < 8; ++j) {
    int k0 = lo + j, k1 = 16 + lo + j;
    float f0 = row[(k0 < kvalid) ? k0 : 0];
    float f1 = row[(k1 < kvalid) ? k1 : 0];
    a[j]     = (_Float16)((mok && k0 < kvalid) ? f0 : 0.0f);
    a[j + 8] = (_Float16)((mok && k1 < kvalid) ? f1 : 0.0f);
  }
  return a;
}

// softmax across the 16 columns of a C-layout tile
__device__ __forceinline__ v8f softmax_tile_cols(v8f s, int ncols, float scale, int lane) {
  int col = lane & 15;
#pragma unroll
  for (int i = 0; i < 8; ++i) {
    float v = (col < ncols) ? s[i] * scale : -1e30f;
    float mx = v;
    mx = fmaxf(mx, __shfl_xor(mx, 1));
    mx = fmaxf(mx, __shfl_xor(mx, 2));
    mx = fmaxf(mx, __shfl_xor(mx, 4));
    mx = fmaxf(mx, __shfl_xor(mx, 8));
    float e = (col < ncols) ? __expf(v - mx) : 0.0f;
    float sum = e;
    sum += __shfl_xor(sum, 1);
    sum += __shfl_xor(sum, 2);
    sum += __shfl_xor(sum, 4);
    sum += __shfl_xor(sum, 8);
    s[i] = e / sum;
  }
  return s;
}

// ---------------- f32 -> f16 conversion (vector 8/thread) ----------------
__global__ void __launch_bounds__(256)
f32_to_f16_kernel(const float* __restrict__ src, _Float16* __restrict__ dst, long n) {
  long i = ((long)blockIdx.x * 256 + threadIdx.x) * 8;
  if (i + 8 <= n) {
    float4 x0 = *reinterpret_cast<const float4*>(src + i);
    float4 x1 = *reinterpret_cast<const float4*>(src + i + 4);
    v8h h;
    h[0] = (_Float16)x0.x; h[1] = (_Float16)x0.y;
    h[2] = (_Float16)x0.z; h[3] = (_Float16)x0.w;
    h[4] = (_Float16)x1.x; h[5] = (_Float16)x1.y;
    h[6] = (_Float16)x1.z; h[7] = (_Float16)x1.w;
    *reinterpret_cast<v8h*>(dst + i) = h;
  }
}

// ---------------- f16-operand GEMM: C[M x Nout] = X[M x K] * W[Nout x K]^T ----------------
// hot loop: 4x b128 loads + 2 shufflevectors + 1 wmma per 32-wide K step
__global__ void __launch_bounds__(128)
gemm_hh_kernel(const _Float16* __restrict__ X, int ldx,
               const _Float16* __restrict__ W, int ldw,
               const float* __restrict__ bias,
               float* __restrict__ C, int ldc, int ccol0,
               int M, int Nout, int K) {
  int lane = threadIdx.x & 31;
  int wave = threadIdx.x >> 5;
  int m0 = blockIdx.x * 16;
  int n0 = blockIdx.y * 64 + wave * 16;
  if (n0 >= Nout) return;
  v8f acc = {};
#pragma unroll 2
  for (int k0 = 0; k0 < K; k0 += 32) {
    __builtin_prefetch(X + (size_t)(m0 + (lane & 15)) * ldx + k0 + 256, 0, 1);
    v16h a = afrag_h(X, ldx, m0, k0, lane);
    v16h b = bfrag_h(W, ldw, n0, k0, lane);
    acc = wmma_f16(a, b, acc);
  }
  int colL = lane & 15;
  int c = n0 + colL;
  float bv = bias ? bias[c] : 0.0f;
#pragma unroll
  for (int i = 0; i < 8; ++i) {
    int r = m0 + ((lane < 16) ? i : 8 + i);
    C[(size_t)r * ldc + ccol0 + c] = acc[i] + bv;
  }
}

// ---------------- spatial differential attention ----------------
__global__ void __launch_bounds__(32)
spatial_diff_kernel(const float* __restrict__ qs, const float* __restrict__ ksb,
                    const float* __restrict__ vsb,
                    const float* __restrict__ lq1, const float* __restrict__ lk1,
                    const float* __restrict__ lq2, const float* __restrict__ lk2,
                    const float* __restrict__ subln, float* __restrict__ st) {
  __shared__ __align__(16) float S[16][cN];       // 32 KB score strip
  int lane = threadIdx.x & 31;
  int n0 = blockIdx.x * 16;
  int g = blockIdx.y;
  int h = g % cH, bt = g / cH;
  const float* qb = qs  + (size_t)bt * cN * 64;
  const float* kb = ksb + (size_t)bt * cN * 64;
  const float* vb = vsb + (size_t)bt * cN * 64;
  const float scale = 0.35355339059f;   // 1/sqrt(8)
  int colL = lane & 15;
  v8f o[2];
  for (int m = 0; m < 2; ++m) {
    __syncthreads();
    int fo = h * 16 + m * cHDS;
    v16h aq = afrag_k8_v(qb + fo, 64, n0, lane);
    for (int kt = 0; kt < cN / 16; ++kt) {
      v16h bk = bfrag_k8_v(kb + fo, 64, kt * 16, lane);
      v8f sacc = {};
      sacc = wmma_f16(aq, bk, sacc);
#pragma unroll
      for (int i = 0; i < 8; ++i) {
        int r = (lane < 16) ? i : 8 + i;
        S[r][kt * 16 + colL] = sacc[i] * scale;
      }
    }
    __syncthreads();
    if (lane < 16) {                       // serial row softmax
      float mx = -1e30f;
      for (int c = 0; c < cN; ++c) mx = fmaxf(mx, S[lane][c]);
      float sum = 0.0f;
      for (int c = 0; c < cN; ++c) { float e = __expf(S[lane][c] - mx); S[lane][c] = e; sum += e; }
      float inv = 1.0f / sum;
      for (int c = 0; c < cN; ++c) S[lane][c] *= inv;
    }
    __syncthreads();
    v8f oa = {};
    for (int k0 = 0; k0 < cN; k0 += 32) {
      v16h ap = afrag_rows_v(&S[0][0], cN, 0, k0, lane);
      v16h bv = bfrag_cols_v(vb + h * 16, 64, k0, 0, lane);
      oa = wmma_f16(ap, bv, oa);
    }
    o[m] = oa;
  }
  float l1 = 0.0f, l2 = 0.0f;
#pragma unroll
  for (int j = 0; j < cHDS; ++j) { l1 += lq1[j] * lk1[j]; l2 += lq2[j] * lk2[j]; }
  float lam = __expf(l1) - __expf(l2) + LAMBDA_INIT;
  float sw = subln[colL];
#pragma unroll
  for (int i = 0; i < 8; ++i) {
    float v = o[0][i] - lam * o[1][i];
    float ss = v * v;
    ss += __shfl_xor(ss, 1);
    ss += __shfl_xor(ss, 2);
    ss += __shfl_xor(ss, 4);
    ss += __shfl_xor(ss, 8);
    float ms = ss * (1.0f / 16.0f);
    v = v * rsqrtf(ms + 1e-6f) * sw * (1.0f - LAMBDA_INIT);
    int r = n0 + ((lane < 16) ? i : 8 + i);
    st[((size_t)bt * cN + r) * 256 + h * 16 + colL] = v;
  }
}

// ---------------- dtw-aggregated spatial attention ----------------
__global__ void __launch_bounds__(32)
agg_scores_kernel(const float* __restrict__ q, const float* __restrict__ k,
                  float* __restrict__ sc, int C) {
  int lane = threadIdx.x & 31;
  int g = blockIdx.z;
  int h = g % cH, bt = g / cH;
  const float* qb = q + (size_t)bt * C * 64 + h * 16;
  const float* kb = k + (size_t)bt * C * 64 + h * 16;
  int m0 = blockIdx.x * 16, n0 = blockIdx.y * 16;
  v16h a = afrag_k16_v(qb, 64, m0, C, lane);
  v16h b = bfrag_k16_v(kb, 64, n0, C, lane);
  v8f s = {};
  s = wmma_f16(a, b, s);
  float* outp = sc + (size_t)g * C * C;
  int colL = lane & 15;
#pragma unroll
  for (int i = 0; i < 8; ++i) {
    int r = m0 + ((lane < 16) ? i : 8 + i);
    outp[(size_t)r * C + n0 + colL] = s[i] * 0.25f;   // 1/sqrt(16)
  }
}

__global__ void __launch_bounds__(32)
agg_apply_kernel(const float* __restrict__ cmap, const float* __restrict__ sc,
                 const float* __restrict__ v, float* __restrict__ sg,
                 int C, int accumulate) {
  __shared__ __align__(16) float P[16][cC1];
  int lane = threadIdx.x & 31;
  int g = blockIdx.y;
  int h = g % cH, bt = g / cH;
  int n0 = blockIdx.x * 16;
  const float* scg = sc + (size_t)g * C * C;
  const float* vb  = v + (size_t)bt * C * 64 + h * 16;
  int colL = lane & 15;
  for (int mt = 0; mt < C / 16; ++mt) {
    v8f p = {};
    for (int k0 = 0; k0 < C; k0 += 32) {
      v16h a = afrag_cols_v(cmap, cN, n0, k0, lane);
      v16h b = bfrag_cols_v(scg, C, k0, mt * 16, lane);
      p = wmma_f16(a, b, p);
    }
#pragma unroll
    for (int i = 0; i < 8; ++i) {
      int r = (lane < 16) ? i : 8 + i;
      P[r][mt * 16 + colL] = p[i];
    }
  }
  __syncthreads();
  if (lane < 16) {
    float mx = -1e30f;
    for (int c = 0; c < C; ++c) mx = fmaxf(mx, P[lane][c]);
    float sum = 0.0f;
    for (int c = 0; c < C; ++c) { float e = __expf(P[lane][c] - mx); P[lane][c] = e; sum += e; }
    float inv = 1.0f / sum;
    for (int c = 0; c < C; ++c) P[lane][c] *= inv;
  }
  __syncthreads();
  v8f o = {};
  for (int k0 = 0; k0 < C; k0 += 32) {
    v16h a = afrag_rows_v(&P[0][0], cC1, 0, k0, lane);
    v16h b = bfrag_cols_v(vb, 64, k0, 0, lane);
    o = wmma_f16(a, b, o);
  }
#pragma unroll
  for (int i = 0; i < 8; ++i) {
    int r = n0 + ((lane < 16) ? i : 8 + i);
    size_t idx = ((size_t)bt * cN + r) * 64 + h * 16 + colL;
    if (accumulate) sg[idx] += o[i]; else sg[idx] = o[i];
  }
}

// ---------------- temporal attention -> st[...,128:192] ----------------
__global__ void __launch_bounds__(32)
temporal_kernel(const float* __restrict__ qt, const float* __restrict__ kt,
                const float* __restrict__ vt, float* __restrict__ st) {
  __shared__ __align__(16) float P[16][16];
  int lane = threadIdx.x & 31;
  int g = blockIdx.x;
  int h = g % cH, bn = g / cH;
  int n = bn % cN, b = bn / cN;
  size_t base = ((size_t)b * cT * cN + n) * 64 + h * 16;
  const int ld = cN * 64;
  int colL = lane & 15;
  v16h a  = afrag_k16_v(qt + base, ld, 0, cT, lane);
  v16h bk = bfrag_k16_v(kt + base, ld, 0, cT, lane);
  v8f s = {};
  s = wmma_f16(a, bk, s);
  s = softmax_tile_cols(s, cT, 0.25f, lane);
#pragma unroll
  for (int i = 0; i < 8; ++i) {
    int r = (lane < 16) ? i : 8 + i;
    P[r][colL] = s[i];
  }
  __syncthreads();
  v16h ap = afrag_k16_v(&P[0][0], 16, 0, 16, lane);
  v16h bv = bfrag_cols_g(vt + base, ld, 0, 0, cT, lane);
  v8f o = {};
  o = wmma_f16(ap, bv, o);
#pragma unroll
  for (int i = 0; i < 8; ++i) {
    int t = (lane < 16) ? i : 8 + i;
    if (t < cT)
      st[(((size_t)b * cT + t) * cN + n) * 256 + 128 + h * 16 + colL] = o[i];
  }
}

// ---------------- aggregative temporal attention -> st[...,192:256] ----------------
__global__ void __launch_bounds__(32)
temporal_agg_kernel(const float* __restrict__ tq, const float* __restrict__ ka,
                    const float* __restrict__ va, const float* __restrict__ tinf,
                    float* __restrict__ st) {
  __shared__ __align__(16) float P[16][16];
  int lane = threadIdx.x & 31;
  int g = blockIdx.x;
  int h = g % cH, bn = g / cH;
  int n = bn % cN, b = bn / cN;
  size_t kvbase = ((size_t)b * cT * cN + n) * 64 + h * 16;
  const int ld = cN * 64;
  int colL = lane & 15;
  const float* qb = tq + (size_t)n * cSEG * 64 + h * 16;
  v16h a1 = afrag_k16_v(qb, 64, 0, cSEG, lane);
  v16h b1 = bfrag_k16_v(ka + kvbase, ld, 0, cT, lane);
  v8f s1 = {};
  s1 = wmma_f16(a1, b1, s1);
#pragma unroll
  for (int i = 0; i < 8; ++i) {
    int r = (lane < 16) ? i : 8 + i;
    P[r][colL] = s1[i] * 0.25f;
  }
  __syncthreads();
  const float* tb = tinf + ((size_t)b * cN + n) * cT * cSEG;
  v16h a2 = afrag_rows_g(tb, cSEG, 0, 0, cT, cSEG, lane);
  v16h b2 = bfrag_cols_g(&P[0][0], 16, 0, 0, cSEG, lane);
  v8f s2 = {};
  s2 = wmma_f16(a2, b2, s2);
  s2 = softmax_tile_cols(s2, cT, 1.0f, lane);
  __syncthreads();
#pragma unroll
  for (int i = 0; i < 8; ++i) {
    int r = (lane < 16) ? i : 8 + i;
    P[r][colL] = s2[i];
  }
  __syncthreads();
  v16h a3 = afrag_k16_v(&P[0][0], 16, 0, 16, lane);
  v16h b3 = bfrag_cols_g(va + kvbase, ld, 0, 0, cT, lane);
  v8f o = {};
  o = wmma_f16(a3, b3, o);
#pragma unroll
  for (int i = 0; i < 8; ++i) {
    int t = (lane < 16) ? i : 8 + i;
    if (t < cT)
      st[(((size_t)b * cT + t) * cN + n) * 256 + 192 + h * 16 + colL] = o[i];
  }
}

// ---------------- host orchestration ----------------
extern "C" void kernel_launch(void* const* d_in, const int* in_sizes, int n_in,
                              void* d_out, int out_size, void* d_ws, size_t ws_size,
                              hipStream_t stream) {
  (void)in_sizes; (void)n_in; (void)out_size; (void)ws_size;
  const float* x     = (const float*)d_in[0];
  const float* dtw0  = (const float*)d_in[1];
  const float* dtw1  = (const float*)d_in[2];
  const float* map0  = (const float*)d_in[3];
  const float* map1  = (const float*)d_in[4];
  const float* tinf  = (const float*)d_in[5];
  const float* lq1   = (const float*)d_in[9];
  const float* lk1   = (const float*)d_in[10];
  const float* lq2   = (const float*)d_in[11];
  const float* lk2   = (const float*)d_in[12];
  const float* subln = (const float*)d_in[13];
  const float* b_agg = (const float*)d_in[21];
  const float* tq    = (const float*)d_in[25];
  float* out = (float*)d_out;
  float* ws  = (float*)d_ws;

  const size_t PROJ = (size_t)MTOK * 64;
  const size_t P0 = (size_t)cB * cT * cC0 * 64;
  const size_t P1 = (size_t)cB * cT * cC1 * 64;
  // ---- f32 region ----
  float* qs  = ws;
  float* k_s = qs  + PROJ;
  float* v_s = k_s + PROJ;
  float* qtb = v_s + PROJ;
  float* ktb = qtb + PROJ;
  float* vtb = ktb + PROJ;
  float* kta = vtb + PROJ;
  float* vta = kta + PROJ;
  float* qa0 = vta + PROJ;
  float* ka0 = qa0 + P0;
  float* va0 = ka0 + P0;
  float* qa1 = va0 + P0;
  float* ka1 = qa1 + P1;
  float* va1 = ka1 + P1;
  float* sc0 = va1 + P1;
  float* sc1 = sc0 + (size_t)GG * cC0 * cC0;
  float* sgr = sc1 + (size_t)GG * cC1 * cC1;
  float* st  = sgr + PROJ;
  // ---- f16 region (after st) ----
  _Float16* hb = (_Float16*)(st + (size_t)MTOK * 256);
  const size_t WSZ = (size_t)64 * cD;         // one 64x256 weight
  _Float16* xh   = hb;                        hb += (size_t)MTOK * cD;
  _Float16* d0h  = hb;                        hb += (size_t)cB * cT * cC0 * cD;
  _Float16* d1h  = hb;                        hb += (size_t)cB * cT * cC1 * cD;
  _Float16* wxh  = hb;                        hb += 8 * WSZ;   // Wq_s,Wk_s,Wv_s,Wq_t,Wk_t,Wv_t,Wk_ta,Wv_ta
  _Float16* wah  = hb;                        hb += 6 * WSZ;   // Wq_a0..Wv_a1
  _Float16* wgh  = hb;                        hb += (size_t)64 * 64;
  _Float16* woh  = hb;                        hb += (size_t)cD * cD;
  _Float16* sgrh = hb;                        hb += PROJ;
  _Float16* sth  = hb;                        hb += (size_t)MTOK * 256;

  auto cvt = [&](const float* src, _Float16* dst, long n) {
    long blocks = (n / 8 + 255) / 256;
    f32_to_f16_kernel<<<dim3((unsigned)blocks), dim3(256), 0, stream>>>(src, dst, n);
  };
  auto gemm = [&](const _Float16* X, int ldx, const _Float16* W, const float* bias,
                  float* C, int ldc, int ccol, int M, int Nout, int K) {
    dim3 grid(M / 16, (Nout + 63) / 64);
    gemm_hh_kernel<<<grid, dim3(128), 0, stream>>>(X, ldx, W, K, bias, C, ldc, ccol, M, Nout, K);
  };

  // ---- one-time f16 conversions of inputs & weights ----
  cvt(x,    xh,  (long)MTOK * cD);
  cvt(dtw0, d0h, (long)cB * cT * cC0 * cD);
  cvt(dtw1, d1h, (long)cB * cT * cC1 * cD);
  const int wx_idx[8] = {6, 7, 8, 22, 23, 24, 26, 27};   // Wq_s..Wv_s, Wq_t..Wv_t, Wk_ta, Wv_ta
  for (int i = 0; i < 8; ++i)
    cvt((const float*)d_in[wx_idx[i]], wxh + (size_t)i * WSZ, (long)WSZ);
  const int wa_idx[6] = {14, 15, 16, 17, 18, 19};
  for (int i = 0; i < 6; ++i)
    cvt((const float*)d_in[wa_idx[i]], wah + (size_t)i * WSZ, (long)WSZ);
  cvt((const float*)d_in[20], wgh, 64 * 64);
  cvt((const float*)d_in[28], woh, (long)cD * cD);

  // ---- projections (pure-f16 GEMMs, f32 out) ----
  float* projs[8] = {qs, k_s, v_s, qtb, ktb, vtb, kta, vta};
  for (int i = 0; i < 8; ++i)
    gemm(xh, cD, wxh + (size_t)i * WSZ, nullptr, projs[i], 64, 0, MTOK, 64, cD);
  const int M0 = cB * cT * cC0, M1 = cB * cT * cC1;
  gemm(d0h, cD, wah + 0 * WSZ, nullptr, qa0, 64, 0, M0, 64, cD);
  gemm(d0h, cD, wah + 1 * WSZ, nullptr, ka0, 64, 0, M0, 64, cD);
  gemm(d0h, cD, wah + 2 * WSZ, nullptr, va0, 64, 0, M0, 64, cD);
  gemm(d1h, cD, wah + 3 * WSZ, nullptr, qa1, 64, 0, M1, 64, cD);
  gemm(d1h, cD, wah + 4 * WSZ, nullptr, ka1, 64, 0, M1, 64, cD);
  gemm(d1h, cD, wah + 5 * WSZ, nullptr, va1, 64, 0, M1, 64, cD);

  // ---- spatial differential attention -> st[:,0:64] ----
  spatial_diff_kernel<<<dim3(cN / 16, GG), dim3(32), 0, stream>>>(
      qs, k_s, v_s, lq1, lk1, lq2, lk2, subln, st);

  // ---- dtw-aggregated attention -> sgr, then W_agg mix -> st[:,64:128] ----
  agg_scores_kernel<<<dim3(cC0 / 16, cC0 / 16, GG), dim3(32), 0, stream>>>(qa0, ka0, sc0, cC0);
  agg_apply_kernel<<<dim3(cN / 16, GG), dim3(32), 0, stream>>>(map0, sc0, va0, sgr, cC0, 0);
  agg_scores_kernel<<<dim3(cC1 / 16, cC1 / 16, GG), dim3(32), 0, stream>>>(qa1, ka1, sc1, cC1);
  agg_apply_kernel<<<dim3(cN / 16, GG), dim3(32), 0, stream>>>(map1, sc1, va1, sgr, cC1, 1);
  cvt(sgr, sgrh, (long)PROJ);
  gemm(sgrh, 64, wgh, b_agg, st, 256, 64, MTOK, 64, 64);

  // ---- temporal branches -> st[:,128:192], st[:,192:256] ----
  temporal_kernel<<<dim3(cB * cN * cH), dim3(32), 0, stream>>>(qtb, ktb, vtb, st);
  temporal_agg_kernel<<<dim3(cB * cN * cH), dim3(32), 0, stream>>>(tq, kta, vta, tinf, st);

  // ---- output projection: out = st @ W_out^T ----
  cvt(st, sth, (long)MTOK * 256);
  gemm(sth, 256, woh, nullptr, out, 256, 0, MTOK, 256, 256);
}
